// WindowAttention_62835371540722
// MI455X (gfx1250) — compile-verified, matrix operand order checked by reading, not run on hardware
//
#include <hip/hip_runtime.h>
#include <math.h>

// ---------------- problem constants ----------------
#define DIMC   512
#define NHEAD  16
#define HDIM   32
#define NTOK   49          // tokens per window (7x7)
#define NPAD   64          // padded tokens per window for 16x WMMA tiles
#define NWIN   2048        // windows * batch
#define MTOT   (NWIN * NTOK)   // 100352 total tokens (multiple of 64)

typedef __bf16 bf16_t;
typedef __attribute__((ext_vector_type(16))) __bf16 v16bf;
typedef __attribute__((ext_vector_type(8)))  __bf16 v8bf;
typedef __attribute__((ext_vector_type(8)))  float  v8f;
typedef int v4i_gcc __attribute__((vector_size(16)));   // matches builtin param

#define AS1 __attribute__((address_space(1)))
#define AS3 __attribute__((address_space(3)))

#if __has_builtin(__builtin_amdgcn_global_load_async_to_lds_b128)
#define HAVE_ASYNC_LDS 1
#else
#define HAVE_ASYNC_LDS 0
#endif

// one 16-byte global->LDS copy (async on CDNA5, sync fallback otherwise)
__device__ __forceinline__ void copy_b128(const bf16_t* g, bf16_t* l) {
#if HAVE_ASYNC_LDS
  __builtin_amdgcn_global_load_async_to_lds_b128(
      (AS1 v4i_gcc*)(v4i_gcc*)g, (AS3 v4i_gcc*)(v4i_gcc*)l, 0, 0);
#else
  *(v8bf*)l = *(const v8bf*)g;
#endif
}

__device__ __forceinline__ void wait_async_le3() {
#if HAVE_ASYNC_LDS
#if __has_builtin(__builtin_amdgcn_s_wait_asynccnt)
  __builtin_amdgcn_s_wait_asynccnt(3);
#else
  asm volatile("s_wait_asynccnt 0x3" ::: "memory");
#endif
#endif
}

__device__ __forceinline__ void wait_async_le0() {
#if HAVE_ASYNC_LDS
#if __has_builtin(__builtin_amdgcn_s_wait_asynccnt)
  __builtin_amdgcn_s_wait_asynccnt(0);
#else
  asm volatile("s_wait_asynccnt 0x0" ::: "memory");
#endif
#endif
}

// Load a 16x32 bf16 A-style WMMA fragment for this lane.
// base = pointer to (row_of_this_lane, k0); per ISA layout the lane needs
// K = half*8 + [0..7]  and  K = 16 + half*8 + [0..7]  (half = lane>>4).
__device__ __forceinline__ v16bf load_frag(const bf16_t* base, int half) {
  const v8bf a0 = *(const v8bf*)(base + half * 8);
  const v8bf a1 = *(const v8bf*)(base + 16 + half * 8);
  v16bf r;
#pragma unroll
  for (int i = 0; i < 8; ++i) { r[i] = a0[i]; r[i + 8] = a1[i]; }
  return r;
}

__device__ __forceinline__ v8f wmma_bf16(v16bf a, v16bf b, v8f c) {
  return __builtin_amdgcn_wmma_f32_16x16x32_bf16(false, a, false, b, (short)0, c,
                                                 false, false);
}

// Stage one K-step: A tile 64x32 (4KB) + B tile 128x32 (8KB) into LDS.
// 256 threads -> each thread: 1 A chunk + 2 B chunks = 3 async instrs/wave.
__device__ __forceinline__ void stage_tile(const bf16_t* __restrict__ Ag,
                                           const bf16_t* __restrict__ Bg,
                                           bf16_t* As, bf16_t* Bs, int t) {
  const int r = t >> 2;
  const int c = (t & 3) * 8;
  copy_b128(Ag + (size_t)r * DIMC + c, As + r * 32 + c);
  copy_b128(Bg + (size_t)r * DIMC + c, Bs + r * 32 + c);
  copy_b128(Bg + (size_t)(r + 64) * DIMC + c, Bs + (r + 64) * 32 + c);
}

// ---------------- prep kernels ----------------
__global__ void k_convert_f32_bf16(const float* __restrict__ src,
                                   bf16_t* __restrict__ dst, int n) {
  int i = blockIdx.x * blockDim.x + threadIdx.x;
  int stride = gridDim.x * blockDim.x;
  for (; i < n; i += stride) dst[i] = (bf16_t)src[i];
}

// Un-interleave qkv weights: out column o = h*32+d of matrix c reads source row
// h*96 + d*3 + c.  Wpack layout: [c][o][k], BiasPack: [c][o].
__global__ void k_pack_qkv(const float* __restrict__ qkv_w,
                           const float* __restrict__ qkv_b,
                           bf16_t* __restrict__ Wpack,
                           float* __restrict__ BiasPack) {
  int i = blockIdx.x * blockDim.x + threadIdx.x;
  int stride = gridDim.x * blockDim.x;
  const int total = 3 * DIMC * DIMC;
  for (; i < total; i += stride) {
    int c = i / (DIMC * DIMC);
    int rem = i - c * (DIMC * DIMC);
    int o = rem / DIMC;
    int k = rem - o * DIMC;
    int h = o >> 5, d = o & 31;
    int srow = h * 96 + d * 3 + c;
    Wpack[i] = (bf16_t)qkv_w[srow * DIMC + k];
    if (k == 0) BiasPack[c * DIMC + o] = qkv_b[srow];
  }
}

// ---------------- fused QKV GEMM (LDS-staged, double buffered) ----------------
// grid: (MTOT/64, 1536/128) ; block 256 (8 waves).
// Block tile 64M x 128N; wave w owns subtiles (tm = w&3, tn = (w>>2)*4 + j).
// Output scattered to attention layouts:
//   Q,K : [b][h][64][32] bf16 (token rows, padded)
//   V   : [b][h][32][64] bf16 (transposed: head-dim rows, padded keys)
__global__ __launch_bounds__(256) void k_qkv_gemm(
    const bf16_t* __restrict__ Xb, const bf16_t* __restrict__ Wpack,
    const float* __restrict__ BiasPack, bf16_t* __restrict__ Qp,
    bf16_t* __restrict__ Kp, bf16_t* __restrict__ Vt) {
  __shared__ bf16_t As[2][64 * 32];
  __shared__ bf16_t Bs[2][128 * 32];

  const int t = threadIdx.x;
  const int w = t >> 5, lane = t & 31;
  const int half = lane >> 4, l15 = lane & 15;
  const int tm = w & 3, tnb = (w >> 2) * 4;

  const int rowBase = blockIdx.x * 64;
  const int colBase = blockIdx.y * 128;
  const bf16_t* Agb = Xb + (size_t)rowBase * DIMC;
  const bf16_t* Bgb = Wpack + (size_t)colBase * DIMC;

  v8f acc[4] = {};
  stage_tile(Agb, Bgb, As[0], Bs[0], t);
#pragma unroll 1
  for (int s = 0; s < DIMC / 32; ++s) {
    const int cur = s & 1;
    if (s + 1 < DIMC / 32) {
      stage_tile(Agb + (s + 1) * 32, Bgb + (s + 1) * 32, As[1 - cur],
                 Bs[1 - cur], t);
      wait_async_le3();
    } else {
      wait_async_le0();
    }
    __syncthreads();
    v16bf a = load_frag(&As[cur][(tm * 16 + l15) * 32], half);
#pragma unroll
    for (int j = 0; j < 4; ++j) {
      v16bf b = load_frag(&Bs[cur][((tnb + j) * 16 + l15) * 32], half);
      acc[j] = wmma_bf16(a, b, acc[j]);
    }
    __syncthreads();
  }

#pragma unroll
  for (int j = 0; j < 4; ++j) {
    int n = colBase + (tnb + j) * 16 + l15;  // fused output column 0..1535
    int mat = n >> 9, o = n & 511;
    int h = o >> 5, d = o & 31;
    float bv = BiasPack[n];
#pragma unroll
    for (int r = 0; r < 8; ++r) {
      int m = rowBase + tm * 16 + r + 8 * half;
      int b_ = m / NTOK;
      int q = m - b_ * NTOK;
      bf16_t val = (bf16_t)(acc[j][r] + bv);
      if (mat == 0)
        Qp[((size_t)(b_ * NHEAD + h) * NPAD + q) * HDIM + d] = val;
      else if (mat == 1)
        Kp[((size_t)(b_ * NHEAD + h) * NPAD + q) * HDIM + d] = val;
      else
        Vt[((size_t)(b_ * NHEAD + h) * HDIM + d) * NPAD + q] = val;
    }
  }
}

// ---------------- fused attention per (window, head) ----------------
// grid: (NWIN, NHEAD), block 256 (8 waves).
__global__ __launch_bounds__(256) void k_attn(
    const bf16_t* __restrict__ Qp, const bf16_t* __restrict__ Kp,
    const bf16_t* __restrict__ Vt, const float* __restrict__ bias_table,
    bf16_t* __restrict__ Ob) {
  __shared__ float  S[NPAD * NPAD];    // 16 KB scores
  __shared__ bf16_t P[NPAD * NPAD];    // 8 KB softmax probs

  const int w = threadIdx.x >> 5, lane = threadIdx.x & 31;
  const int half = lane >> 4, l15 = lane & 15;
  const int bwin = blockIdx.x, h = blockIdx.y;
  const size_t bh = (size_t)bwin * NHEAD + h;

  // ---- phase 1: S = Q @ K^T  (16 tiles, 2 per wave) ----
  const bf16_t* qbase = Qp + bh * NPAD * HDIM;
  const bf16_t* kbase = Kp + bh * NPAD * HDIM;
#pragma unroll
  for (int i = 0; i < 2; ++i) {
    int tile = w * 2 + i;
    int tm = tile >> 2, tn = tile & 3;
    v16bf a = load_frag(qbase + (size_t)(tm * 16 + l15) * HDIM, half);
    v16bf b = load_frag(kbase + (size_t)(tn * 16 + l15) * HDIM, half);
    v8f acc = {};
    acc = wmma_bf16(a, b, acc);
#pragma unroll
    for (int r = 0; r < 8; ++r)
      S[(tm * 16 + r + 8 * half) * NPAD + tn * 16 + l15] = acc[r];
  }
  __syncthreads();

  // ---- phase 2: masked softmax + relative-position bias (8 rows / wave) ----
  const float scale = 0.17677669529663687f;  // 1/sqrt(32)
#pragma unroll
  for (int rr = 0; rr < 8; ++rr) {
    int q = w * 8 + rr;
    bool qv = q < NTOK;
    int qh = q / 7, qw = q - qh * 7;
    int c0 = lane, c1 = lane + 32;
    bool v0ok = qv && (c0 < NTOK);
    bool v1ok = qv && (c1 < NTOK);
    float v0 = -__builtin_inff(), v1 = -__builtin_inff();
    if (v0ok) {
      int ch = c0 / 7, cw = c0 - ch * 7;
      v0 = S[q * NPAD + c0] * scale +
           bias_table[((qh - ch + 6) * 13 + (qw - cw + 6)) * NHEAD + h];
    }
    if (v1ok) {
      int ch = c1 / 7, cw = c1 - ch * 7;
      v1 = S[q * NPAD + c1] * scale +
           bias_table[((qh - ch + 6) * 13 + (qw - cw + 6)) * NHEAD + h];
    }
    float mx = fmaxf(v0, v1);
#pragma unroll
    for (int off = 16; off >= 1; off >>= 1)
      mx = fmaxf(mx, __shfl_xor(mx, off, 32));
    float e0 = v0ok ? __expf(v0 - mx) : 0.f;
    float e1 = v1ok ? __expf(v1 - mx) : 0.f;
    float s = e0 + e1;
#pragma unroll
    for (int off = 16; off >= 1; off >>= 1) s += __shfl_xor(s, off, 32);
    float inv = (s > 0.f) ? (1.f / s) : 0.f;
    P[q * NPAD + c0] = (bf16_t)(e0 * inv);
    P[q * NPAD + c1] = (bf16_t)(e1 * inv);
  }
  __syncthreads();

  // ---- phase 3: O = P @ V  (8 tiles, 1 per wave; K-dim = 64) ----
  const bf16_t* vbase = Vt + bh * HDIM * NPAD;
  {
    int tm = w >> 1, tn = w & 1;
    v8f acc = {};
#pragma unroll
    for (int k0 = 0; k0 < NPAD; k0 += 32) {
      v16bf a = load_frag(&P[(tm * 16 + l15) * NPAD + k0], half);
      v16bf b = load_frag(vbase + (size_t)(tn * 16 + l15) * NPAD + k0, half);
      acc = wmma_bf16(a, b, acc);
    }
#pragma unroll
    for (int r = 0; r < 8; ++r) {
      int q = tm * 16 + r + 8 * half;
      if (q < NTOK) {
        int d = tn * 16 + l15;
        size_t tok = (size_t)bwin * NTOK + q;
        Ob[tok * DIMC + h * HDIM + d] = (bf16_t)acc[r];
      }
    }
  }
}

// ---------------- output projection GEMM (LDS-staged) ----------------
// grid: (MTOT/64, DIMC/128), block 256. f32 output + bias -> d_out.
__global__ __launch_bounds__(256) void k_proj_gemm(
    const bf16_t* __restrict__ Ob, const bf16_t* __restrict__ Pw,
    const float* __restrict__ proj_b, float* __restrict__ out) {
  __shared__ bf16_t As[2][64 * 32];
  __shared__ bf16_t Bs[2][128 * 32];

  const int t = threadIdx.x;
  const int w = t >> 5, lane = t & 31;
  const int half = lane >> 4, l15 = lane & 15;
  const int tm = w & 3, tnb = (w >> 2) * 4;

  const int rowBase = blockIdx.x * 64;
  const int colBase = blockIdx.y * 128;
  const bf16_t* Agb = Ob + (size_t)rowBase * DIMC;
  const bf16_t* Bgb = Pw + (size_t)colBase * DIMC;

  v8f acc[4] = {};
  stage_tile(Agb, Bgb, As[0], Bs[0], t);
#pragma unroll 1
  for (int s = 0; s < DIMC / 32; ++s) {
    const int cur = s & 1;
    if (s + 1 < DIMC / 32) {
      stage_tile(Agb + (s + 1) * 32, Bgb + (s + 1) * 32, As[1 - cur],
                 Bs[1 - cur], t);
      wait_async_le3();
    } else {
      wait_async_le0();
    }
    __syncthreads();
    v16bf a = load_frag(&As[cur][(tm * 16 + l15) * 32], half);
#pragma unroll
    for (int j = 0; j < 4; ++j) {
      v16bf b = load_frag(&Bs[cur][((tnb + j) * 16 + l15) * 32], half);
      acc[j] = wmma_bf16(a, b, acc[j]);
    }
    __syncthreads();
  }

#pragma unroll
  for (int j = 0; j < 4; ++j) {
    int n = colBase + (tnb + j) * 16 + l15;
    float bv = proj_b[n];
#pragma unroll
    for (int r = 0; r < 8; ++r) {
      int m = rowBase + tm * 16 + r + 8 * half;
      out[(size_t)m * DIMC + n] = acc[j][r] + bv;
    }
  }
}

// ---------------- host launcher ----------------
extern "C" void kernel_launch(void* const* d_in, const int* in_sizes, int n_in,
                              void* d_out, int out_size, void* d_ws,
                              size_t ws_size, hipStream_t stream) {
  const float* x      = (const float*)d_in[0];
  const float* qkv_w  = (const float*)d_in[1];
  const float* qkv_b  = (const float*)d_in[2];
  const float* proj_w = (const float*)d_in[3];
  const float* proj_b = (const float*)d_in[4];
  const float* btab   = (const float*)d_in[5];

  char* ws = (char*)d_ws;
  const size_t szXb   = (size_t)MTOT * DIMC * 2;            // x in bf16
  const size_t szWp   = (size_t)3 * DIMC * DIMC * 2;        // packed qkv W
  const size_t szBp   = (size_t)3 * DIMC * 4;               // packed qkv bias
  const size_t szPw   = (size_t)DIMC * DIMC * 2;            // proj W bf16
  const size_t szQKV  = (size_t)NWIN * NHEAD * NPAD * HDIM * 2;  // per matrix
  const size_t szOb   = (size_t)MTOT * DIMC * 2;            // attn out bf16

  size_t off = 0;
  bf16_t* Xb = (bf16_t*)(ws + off); off += szXb;
  bf16_t* Wp = (bf16_t*)(ws + off); off += szWp;
  float*  Bp = (float*)(ws + off);  off += szBp;
  bf16_t* Pw = (bf16_t*)(ws + off); off += szPw;
  bf16_t* Qp = (bf16_t*)(ws + off); off += szQKV;
  bf16_t* Kp = (bf16_t*)(ws + off); off += szQKV;
  bf16_t* Vt = (bf16_t*)(ws + off); off += szQKV;
  bf16_t* Ob = (bf16_t*)(ws + off); off += szOb;
  (void)ws_size; (void)in_sizes; (void)n_in; (void)out_size;

  // zero the padded Q/K/V region so pad rows can never inject NaN/Inf
  (void)hipMemsetAsync((void*)Qp, 0, 3 * szQKV, stream);

  k_convert_f32_bf16<<<8192, 256, 0, stream>>>(x, Xb, MTOT * DIMC);
  k_pack_qkv<<<3072, 256, 0, stream>>>(qkv_w, qkv_b, Wp, Bp);
  k_convert_f32_bf16<<<1024, 256, 0, stream>>>(proj_w, Pw, DIMC * DIMC);

  dim3 g2(MTOT / 64, (3 * DIMC) / 128);
  k_qkv_gemm<<<g2, 256, 0, stream>>>(Xb, Wp, Bp, Qp, Kp, Vt);

  dim3 g3(NWIN, NHEAD);
  k_attn<<<g3, 256, 0, stream>>>(Qp, Kp, Vt, btab, Ob);

  dim3 g4(MTOT / 64, DIMC / 128);
  k_proj_gemm<<<g4, 256, 0, stream>>>(Ob, Pw, proj_b, (float*)d_out);
}
